// CausalAttention_57389353009417
// MI455X (gfx1250) — matrix operand order repaired and checked by reading, not compiled, not run on hardware
//
#include <hip/hip_runtime.h>
#include <hip/hip_bf16.h>
#include <cstdint>

// Problem constants (match reference)
#define B_   4
#define T_   2048
#define D_   1024
#define H_   16
#define HD_  64

typedef __bf16 v16bf  __attribute__((ext_vector_type(16)));
typedef __bf16 bf16x8 __attribute__((ext_vector_type(8)));
typedef float  v8f    __attribute__((ext_vector_type(8)));
typedef unsigned int v4u __attribute__((ext_vector_type(4)));
typedef int    v4i    __attribute__((ext_vector_type(4)));
typedef int    v8i    __attribute__((ext_vector_type(8)));

#define WMMA_BF16(a, b, c) \
  __builtin_amdgcn_wmma_f32_16x16x32_bf16(false, (a), false, (b), (short)0, (c), false, false)

// ---------------------------------------------------------------------------
// Fragment loaders (wave32, 16x16x32 bf16 WMMA; contiguous k = 16*half + s, same in A and B)
//
// A (16x32, row-major source, stride lda):
//   lanes 0-15 : row=lane,    elems[0..15]=K 0..15
//   lanes 16-31: row=lane-16, elems[0..15]=K 16..31
static __device__ __forceinline__ v16bf load_a_frag(const __bf16* A, int lda,
                                                    int k0, int lane) {
  const int row = lane & 15, half = lane >> 4;
  const __bf16* p = A + (size_t)row * lda + k0 + half * 16;
  bf16x8 lo = *(const bf16x8*)(p);
  bf16x8 hi = *(const bf16x8*)(p + 8);
  v16bf f;
#pragma unroll
  for (int i = 0; i < 8; ++i) { f[i] = lo[i]; f[8 + i] = hi[i]; }
  return f;
}

// B (32x16 KxN, source stored column-major: Bt[n*ldb + k]):
//   lanes 0-15 : col N=lane,    K = k0+0..15
//   lanes 16-31: col N=lane-16, K = k0+16..31
static __device__ __forceinline__ v16bf load_b_frag(const __bf16* Bt, int ldb,
                                                    int n0, int k0, int lane) {
  const int col = lane & 15, half = lane >> 4;
  return *(const v16bf*)(Bt + (size_t)(n0 + col) * ldb + k0 + half * 16);
}

// C/D (16x16 f32): VGPR r, lanes 0-15 -> (M=r, N=lane); lanes 16-31 -> (M=r+8)

// ---------------------------------------------------------------------------
// Tensor Data Mover: async 2-D bf16 tile load global -> LDS.
// D# built per CDNA5 ISA 8.3-8.6: group0 = {count=1 | lds_addr | global_addr |
// type=2}, group1 = {data_size=2B, tensor dims == tile dims, dim0 stride}.
// Issued by one wave; tracked with TENSORcnt.
static __device__ __forceinline__ void tdm_load_2d_bf16(unsigned lds_byte_addr,
                                                        const __bf16* gsrc,
                                                        unsigned d0, unsigned d1,
                                                        unsigned long long stride0) {
  const unsigned long long ga = (unsigned long long)(uintptr_t)gsrc;
  v4u g0;
  g0[0] = 1u;                                        // count=1 (valid user D#)
  g0[1] = lds_byte_addr;                             // LDS destination (bytes)
  g0[2] = (unsigned)ga;                              // global_addr[31:0]
  g0[3] = (unsigned)((ga >> 32) & 0x01ffffffull) | (2u << 30);  // addr[56:32], type=2
  v8i g1;
  g1[0] = 0x00010000;                                // wg_mask=0, data_size=1 (2B)
  g1[1] = (int)((d0 & 0xffffu) << 16);               // tensor_dim0[15:0]
  g1[2] = (int)(((d0 >> 16) & 0xffffu) | ((d1 & 0xffffu) << 16));  // td0 hi | td1 lo
  g1[3] = (int)(((d1 >> 16) & 0xffffu) | ((d0 & 0xffffu) << 16));  // td1 hi | tile_dim0
  g1[4] = (int)(d1 & 0xffffu);                       // tile_dim1 (tile_dim2=0)
  g1[5] = (int)(unsigned)(stride0 & 0xffffffffull);  // tensor_dim0_stride[31:0]
  g1[6] = (int)(unsigned)((stride0 >> 32) & 0xffffull);
  g1[7] = 0;
  const v4i z4 = {0, 0, 0, 0};
#if __clang_major__ >= 23
  const v8i z8 = {0, 0, 0, 0, 0, 0, 0, 0};
  __builtin_amdgcn_tensor_load_to_lds(g0, g1, z4, z4, z8, 0);
#else
  __builtin_amdgcn_tensor_load_to_lds(g0, g1, z4, z4, 0);
#endif
}

// ---------------------------------------------------------------------------
// Conversions
__global__ void cvt_bf16(const float* __restrict__ s, __bf16* __restrict__ d, int n) {
  int i = blockIdx.x * blockDim.x + threadIdx.x;
  if (i < n) d[i] = (__bf16)s[i];
}

// W[k*N + n] (row-major KxN) -> Wt[n*K + k] (column-major), bf16
__global__ void cvt_transpose(const float* __restrict__ W, __bf16* __restrict__ Wt,
                              int K, int N) {
  int i = blockIdx.x * blockDim.x + threadIdx.x;
  if (i < K * N) {
    int n = i / K, k = i - n * K;
    Wt[(size_t)n * K + k] = (__bf16)W[(size_t)k * N + n];
  }
}

// ---------------------------------------------------------------------------
// Shared GEMM body: C[32x64 per wave] = A[M x K] * Bt[N x K]^T, bf16->f32
template <int KDIM>
static __device__ __forceinline__ void gemm_wave_tile(const __bf16* __restrict__ A,
                                                      const __bf16* __restrict__ Bt,
                                                      int m0, int n0, int lane,
                                                      v8f acc[2][4]) {
#pragma unroll 1
  for (int kk = 0; kk < KDIM; kk += 32) {
    if (kk + 32 < KDIM) {  // hint the next k-slice into cache (global_prefetch_b8)
      __builtin_prefetch(A + (size_t)m0 * KDIM + kk + 32, 0, 1);
      __builtin_prefetch(Bt + (size_t)n0 * KDIM + kk + 32, 0, 1);
    }
    v16bf a0 = load_a_frag(A + (size_t)m0 * KDIM, KDIM, kk, lane);
    v16bf a1 = load_a_frag(A + (size_t)(m0 + 16) * KDIM, KDIM, kk, lane);
    v16bf b0 = load_b_frag(Bt, KDIM, n0 +  0, kk, lane);
    v16bf b1 = load_b_frag(Bt, KDIM, n0 + 16, kk, lane);
    v16bf b2 = load_b_frag(Bt, KDIM, n0 + 32, kk, lane);
    v16bf b3 = load_b_frag(Bt, KDIM, n0 + 48, kk, lane);
    acc[0][0] = WMMA_BF16(a0, b0, acc[0][0]);
    acc[0][1] = WMMA_BF16(a0, b1, acc[0][1]);
    acc[0][2] = WMMA_BF16(a0, b2, acc[0][2]);
    acc[0][3] = WMMA_BF16(a0, b3, acc[0][3]);
    acc[1][0] = WMMA_BF16(a1, b0, acc[1][0]);
    acc[1][1] = WMMA_BF16(a1, b1, acc[1][1]);
    acc[1][2] = WMMA_BF16(a1, b2, acc[1][2]);
    acc[1][3] = WMMA_BF16(a1, b3, acc[1][3]);
  }
}

// ---------------------------------------------------------------------------
// Kernel 1: qkv = x @ W_qkv + b_qkv; scatter into Q,K [B,H,T,64] and Vt [B,H,64,T]
__global__ __launch_bounds__(256) void qkv_gemm(const __bf16* __restrict__ xb,
                                                const __bf16* __restrict__ wt,
                                                const float* __restrict__ bqkv,
                                                __bf16* __restrict__ Qb,
                                                __bf16* __restrict__ Kb,
                                                __bf16* __restrict__ Vt) {
  const int tid = threadIdx.x, lane = tid & 31, w = tid >> 5;
  const int half = lane >> 4, cl = lane & 15;
  const int wm = w & 3, wn = w >> 2;
  const int m0 = blockIdx.x * 128 + wm * 32;      // rows: b*T + t
  const int n0 = blockIdx.y * 128 + wn * 64;      // cols: 0..3071

  v8f acc[2][4];
#pragma unroll
  for (int i = 0; i < 2; ++i)
#pragma unroll
    for (int j = 0; j < 4; ++j)
#pragma unroll
      for (int r = 0; r < 8; ++r) acc[i][j][r] = 0.f;

  gemm_wave_tile<D_>(xb, wt, m0, n0, lane, acc);

  // n0 is 64-aligned -> the wave's whole 64-wide N tile is in one of Q/K/V and
  // one head: sel/h/b are wave-uniform (scalar branch, no exec divergence).
  const int sel = n0 >> 10;                 // 0=Q 1=K 2=V
  const int h   = (n0 & 1023) >> 6;
  const int b   = m0 >> 11;                 // whole 128-row block in one batch
  const size_t bh = (size_t)(b * H_ + h);

  if (sel < 2) {
    __bf16* __restrict__ dst = (sel == 0) ? Qb : Kb;
#pragma unroll
    for (int i = 0; i < 2; ++i)
#pragma unroll
      for (int j = 0; j < 4; ++j)
#pragma unroll
        for (int r = 0; r < 8; ++r) {
          const int t  = (m0 + i * 16 + r + 8 * half) & (T_ - 1);
          const int hd = j * 16 + cl;
          const float val = acc[i][j][r] + bqkv[n0 + j * 16 + cl];
          dst[(bh * T_ + t) * HD_ + hd] = (__bf16)val;
        }
  } else {
#pragma unroll
    for (int i = 0; i < 2; ++i)
#pragma unroll
      for (int j = 0; j < 4; ++j)
#pragma unroll
        for (int r = 0; r < 8; ++r) {
          const int t  = (m0 + i * 16 + r + 8 * half) & (T_ - 1);
          const int hd = j * 16 + cl;
          const float val = acc[i][j][r] + bqkv[n0 + j * 16 + cl];
          Vt[(bh * HD_ + hd) * T_ + t] = (__bf16)val;   // transposed
        }
  }
}

// ---------------------------------------------------------------------------
// Kernel 2: flash attention. grid = (T/128, B*H), block = 256 (8 waves).
// K/Vt key-blocks staged into LDS by the Tensor Data Mover (double-buffered);
// each wave owns 16 query rows; online softmax in f32.
__global__ __launch_bounds__(256) void attn_fwd(const __bf16* __restrict__ Qb,
                                                const __bf16* __restrict__ Kb,
                                                const __bf16* __restrict__ Vt,
                                                const float* __restrict__ eb,
                                                __bf16* __restrict__ Ob) {
  __shared__ __align__(16) __bf16 kbuf[2][32 * HD_];   // [key][hd]   4KB x2
  __shared__ __align__(16) __bf16 vbuf[2][HD_ * 32];   // [hd][key]   4KB x2
  __shared__ __align__(16) __bf16 pbuf[8 * 16 * 32];   // per-wave P  8KB

  const int tid = threadIdx.x, lane = tid & 31, w = tid >> 5;
  const int half = lane >> 4, cl = lane & 15;
  const int bh = blockIdx.y, b = bh >> 4, h = bh & 15;
  const int q0 = blockIdx.x * 128 + w * 16;

  const __bf16* Qp = Qb + (size_t)bh * T_ * HD_ + (size_t)q0 * HD_;
  const __bf16* Kp = Kb + (size_t)bh * T_ * HD_;
  const __bf16* Vp = Vt + (size_t)bh * HD_ * T_;
  const float*  ep = eb + (size_t)b * T_ * T_;

  // Q fragments for the two K=32 slices of hd=64 (reused over all key blocks)
  const v16bf qf0 = load_a_frag(Qp, HD_, 0, lane);
  const v16bf qf1 = load_a_frag(Qp, HD_, 32, lane);

  v8f o[4];
#pragma unroll
  for (int t = 0; t < 4; ++t)
#pragma unroll
    for (int r = 0; r < 8; ++r) o[t][r] = 0.f;

  float mrow[8], lrow[8];
#pragma unroll
  for (int r = 0; r < 8; ++r) { mrow[r] = -__builtin_inff(); lrow[r] = 0.f; }

  __bf16* pw = pbuf + w * (16 * 32);
  const int kend = blockIdx.x * 128 + 128;   // uniform across the block
  const float scale = 0.125f;                // hd^-0.5

  // TDM stage of key block kb into buffer `buf` (K: 32x64 contiguous rows;
  // Vt: 64 rows x 32 keys, row stride T_)
  auto stage = [&](int buf, int kb) {
    tdm_load_2d_bf16((unsigned)(uintptr_t)&kbuf[buf][0], Kp + (size_t)kb * HD_,
                     HD_, 32, HD_);
    tdm_load_2d_bf16((unsigned)(uintptr_t)&vbuf[buf][0], Vp + kb,
                     32, HD_, T_);
  };

  if (w == 0) stage(0, 0);

#pragma unroll 1
  for (int kb = 0; kb < kend; kb += 32) {
    const int buf = (kb >> 5) & 1;
    if (w == 0) {
      if (kb + 32 < kend) {
        stage(buf ^ 1, kb + 32);                    // prefetch next block
        __builtin_amdgcn_s_wait_tensorcnt(2);       // current 2 loads done
      } else {
        __builtin_amdgcn_s_wait_tensorcnt(0);
      }
    }
    __syncthreads();   // LDS tiles for this block visible to all waves

    const __bf16* klds = &kbuf[buf][0];
    const __bf16* vlds = &vbuf[buf][0];

    // ---- S = Q K^T for 16 q x 32 k (two 16x16 C tiles, K-dim = hd = 64)
    v8f s0, s1;
#pragma unroll
    for (int r = 0; r < 8; ++r) { s0[r] = 0.f; s1[r] = 0.f; }
    {
      v16bf k00 = load_b_frag(klds, HD_, 0,  0,  lane);
      v16bf k01 = load_b_frag(klds, HD_, 0,  32, lane);
      s0 = WMMA_BF16(qf0, k00, s0);
      s0 = WMMA_BF16(qf1, k01, s0);
      v16bf k10 = load_b_frag(klds, HD_, 16, 0,  lane);
      v16bf k11 = load_b_frag(klds, HD_, 16, 32, lane);
      s1 = WMMA_BF16(qf0, k10, s1);
      s1 = WMMA_BF16(qf1, k11, s1);
    }

    // ---- scale + entity bias + causal mask (C layout: row=r+8*half, col=cl)
    float sv0[8], sv1[8], mx[8];
#pragma unroll
    for (int r = 0; r < 8; ++r) {
      const int q = q0 + r + 8 * half;
      const int k0i = kb + cl, k1i = kb + 16 + cl;
      float v0 = s0[r] * scale + ep[(size_t)q * T_ + k0i];
      float v1 = s1[r] * scale + ep[(size_t)q * T_ + k1i];
      if (k0i > q) v0 = -1e9f;
      if (k1i > q) v1 = -1e9f;
      sv0[r] = v0; sv1[r] = v1;
      mx[r] = fmaxf(v0, v1);
    }
    // row max across the 16 lanes holding one row
#pragma unroll
    for (int off = 1; off < 16; off <<= 1)
#pragma unroll
      for (int r = 0; r < 8; ++r) mx[r] = fmaxf(mx[r], __shfl_xor(mx[r], off, 32));

    // ---- online softmax update; write P (bf16) to LDS in C layout
    float ps[8];
#pragma unroll
    for (int r = 0; r < 8; ++r) {
      const float mn = fmaxf(mrow[r], mx[r]);
      const float corr = __expf(mrow[r] - mn);
      mrow[r] = mn;
      const float p0 = __expf(sv0[r] - mn);
      const float p1 = __expf(sv1[r] - mn);
      ps[r] = p0 + p1;
      lrow[r] *= corr;
#pragma unroll
      for (int t = 0; t < 4; ++t) o[t][r] *= corr;
      const int row = r + 8 * half;
      pw[row * 32 + cl]      = (__bf16)p0;
      pw[row * 32 + 16 + cl] = (__bf16)p1;
    }
#pragma unroll
    for (int off = 1; off < 16; off <<= 1)
#pragma unroll
      for (int r = 0; r < 8; ++r) ps[r] += __shfl_xor(ps[r], off, 32);
#pragma unroll
    for (int r = 0; r < 8; ++r) lrow[r] += ps[r];

    // LDS ops are in-order per wave; make stores visible before A-layout reads
    asm volatile("s_wait_dscnt 0" ::: "memory");

    // ---- O += P @ V  (P: 16x32 A-frag from LDS; V tile column-major in LDS)
    const v16bf pf = load_a_frag(pw, 32, 0, lane);
#pragma unroll
    for (int t = 0; t < 4; ++t) {
      v16bf vf = load_b_frag(vlds, 32, t * 16, 0, lane);
      o[t] = WMMA_BF16(pf, vf, o[t]);
    }

    __syncthreads();   // all waves done with this buffer before TDM overwrites
  }

  // ---- normalize and store O as bf16 in [B,T,H,HD] (= [B,T,D] feature order)
#pragma unroll
  for (int r = 0; r < 8; ++r) {
    const int q = q0 + r + 8 * half;
    const float inv = 1.0f / lrow[r];
    const size_t base = ((size_t)(b * T_ + q) * H_ + h) * HD_;
#pragma unroll
    for (int t = 0; t < 4; ++t)
      Ob[base + t * 16 + cl] = (__bf16)(o[t][r] * inv);
  }
}

// ---------------------------------------------------------------------------
// Kernel 3: out = O @ W_out + b_out (f32 output)
__global__ __launch_bounds__(256) void out_gemm(const __bf16* __restrict__ Ab,
                                                const __bf16* __restrict__ wt,
                                                const float* __restrict__ bout,
                                                float* __restrict__ out) {
  const int tid = threadIdx.x, lane = tid & 31, w = tid >> 5;
  const int half = lane >> 4, cl = lane & 15;
  const int wm = w & 3, wn = w >> 2;
  const int m0 = blockIdx.x * 128 + wm * 32;
  const int n0 = blockIdx.y * 128 + wn * 64;

  v8f acc[2][4];
#pragma unroll
  for (int i = 0; i < 2; ++i)
#pragma unroll
    for (int j = 0; j < 4; ++j)
#pragma unroll
      for (int r = 0; r < 8; ++r) acc[i][j][r] = 0.f;

  gemm_wave_tile<D_>(Ab, wt, m0, n0, lane, acc);

#pragma unroll
  for (int i = 0; i < 2; ++i)
#pragma unroll
    for (int j = 0; j < 4; ++j)
#pragma unroll
      for (int r = 0; r < 8; ++r) {
        const int m = m0 + i * 16 + r + 8 * half;
        const int n = n0 + j * 16 + cl;
        out[(size_t)m * D_ + n] = acc[i][j][r] + bout[n];
      }
}

// ---------------------------------------------------------------------------
extern "C" void kernel_launch(void* const* d_in, const int* in_sizes, int n_in,
                              void* d_out, int out_size, void* d_ws, size_t ws_size,
                              hipStream_t stream) {
  const float* x    = (const float*)d_in[0];
  const float* eb   = (const float*)d_in[1];
  const float* Wqkv = (const float*)d_in[2];
  const float* bqkv = (const float*)d_in[3];
  const float* Wout = (const float*)d_in[4];
  const float* bout = (const float*)d_in[5];
  float* out = (float*)d_out;

  char* ws = (char*)d_ws;
  size_t off = 0;
  const size_t NX = (size_t)B_ * T_ * D_;          // 8.39M elems
  __bf16* xb    = (__bf16*)(ws + off); off += NX * 2;             // reused as Ob
  __bf16* wqkvt = (__bf16*)(ws + off); off += (size_t)D_ * 3 * D_ * 2;
  __bf16* woutt = (__bf16*)(ws + off); off += (size_t)D_ * D_ * 2;
  __bf16* Qb    = (__bf16*)(ws + off); off += NX * 2;
  __bf16* Kb    = (__bf16*)(ws + off); off += NX * 2;
  __bf16* Vt    = (__bf16*)(ws + off); off += NX * 2;
  __bf16* Ob    = xb;  // x is dead after qkv_gemm; reuse its slot (~72MB total)

  // 1) precision conversion / weight transposes
  cvt_bf16<<<dim3((unsigned)((NX + 255) / 256)), 256, 0, stream>>>(x, xb, (int)NX);
  cvt_transpose<<<dim3((D_ * 3 * D_ + 255) / 256), 256, 0, stream>>>(Wqkv, wqkvt, D_, 3 * D_);
  cvt_transpose<<<dim3((D_ * D_ + 255) / 256), 256, 0, stream>>>(Wout, woutt, D_, D_);

  // 2) QKV projection
  dim3 g1(B_ * T_ / 128, 3 * D_ / 128);
  qkv_gemm<<<g1, 256, 0, stream>>>(xb, wqkvt, bqkv, Qb, Kb, Vt);

  // 3) fused causal attention with entity bias (TDM-staged K/V)
  dim3 g2(T_ / 128, B_ * H_);
  attn_fwd<<<g2, 256, 0, stream>>>(Qb, Kb, Vt, eb, Ob);

  // 4) output projection
  dim3 g3(B_ * T_ / 128, D_ / 128);
  out_gemm<<<g3, 256, 0, stream>>>(Ob, woutt, bout, out);
}
